// MultiHeadAttention_29119878267522
// MI455X (gfx1250) — matrix operand order repaired and checked by reading, not compiled, not run on hardware
//
#include <hip/hip_runtime.h>

// ---------------------------------------------------------------------------
// MI455X (gfx1250) multi-head attention, bf16 WMMA (v_wmma_f32_16x16x32_bf16)
// ---------------------------------------------------------------------------

#define D_MODEL 1024
#define NUM_HEADS 16
#define D_K 64
#define SEQ 2048
#define BATCH 2
#define MROWS (BATCH * SEQ)          // 4096

typedef __bf16 bf16_t;
typedef __attribute__((ext_vector_type(16))) __bf16 v16bf;
typedef __attribute__((ext_vector_type(8)))  float  v8f;
typedef __attribute__((ext_vector_type(8)))  unsigned short u16x8;
typedef __attribute__((ext_vector_type(4)))  unsigned short u16x4;
typedef __attribute__((ext_vector_type(4)))  float  f32x4;

union V16 {                          // bf16 fragment viewed as 2x 16B halves
    v16bf  bf;
    u16x8  h[2];
};

__device__ __forceinline__ unsigned short f2bf(float f) {
    union { float f; unsigned u; } x; x.f = f;
    unsigned r = x.u + 0x7fffu + ((x.u >> 16) & 1u);   // round-to-nearest-even
    return (unsigned short)(r >> 16);
}

__device__ __forceinline__ v8f wmma_bf16(v16bf a, v16bf b, v8f c) {
    // (neg_a, A, neg_b, B, c_mod, C, reuse_a, reuse_b)
    return __builtin_amdgcn_wmma_f32_16x16x32_bf16(false, a, false, b,
                                                   (short)0, c, false, false);
}

// ---------------------------------------------------------------------------
// Projection GEMM: out[m,n] = sum_k A[m,k] * W[n,k] + bias[n]
// M=4096, N=1024, K=1024. Block tile 128x128, 8 waves of 32x64, K-step 32.
// MODE 0: A fp32, store bf16 Q head-layout  [b,h,s,d]
// MODE 1: A fp32, store bf16 K head-layout  [b,h,s,d]
// MODE 2: A fp32, store bf16 V transposed   [b,h,d,s]
// MODE 3: A bf16 (ushort), store fp32 [m,n] (final projection)
// ---------------------------------------------------------------------------
template <int MODE>
__global__ __launch_bounds__(256) void proj_gemm(
    const float* __restrict__ Afp, const unsigned short* __restrict__ Abf,
    const float* __restrict__ W,   const float* __restrict__ bias,
    unsigned short* __restrict__ obf, float* __restrict__ ofp)
{
    constexpr int K = D_MODEL, N = D_MODEL;
    constexpr int LDA = 40;                       // padded LDS row (16B-aligned stride)
    __shared__ unsigned short As[128 * LDA];
    __shared__ unsigned short Ws[128 * LDA];

    const int tid  = threadIdx.x;
    const int lane = tid & 31;
    const int wave = tid >> 5;
    const int m0 = blockIdx.y * 128;
    const int n0 = blockIdx.x * 128;
    const int wm = (wave >> 1) * 32;              // wave m-offset within block
    const int wn = (wave & 1) * 64;               // wave n-offset within block

    v8f acc[2][4];
#pragma unroll
    for (int i = 0; i < 2; ++i)
#pragma unroll
        for (int j = 0; j < 4; ++j) acc[i][j] = (v8f)0.0f;

    const int lr = tid >> 3;                      // 0..31 base load row
    const int lc = (tid & 7) * 4;                 // 0..28 load col

    for (int k0 = 0; k0 < K; k0 += 32) {
        __syncthreads();
        // ---- stage A (fp32->bf16 or bf16 copy) and W (fp32->bf16) tiles ----
#pragma unroll
        for (int rr = 0; rr < 128; rr += 32) {
            const int row = lr + rr;
            if (MODE == 3) {
                u16x4 v = *(const u16x4*)(Abf + (size_t)(m0 + row) * K + k0 + lc);
                *(u16x4*)&As[row * LDA + lc] = v;
            } else {
                f32x4 v = *(const f32x4*)(Afp + (size_t)(m0 + row) * K + k0 + lc);
                u16x4 o;
                o[0] = f2bf(v[0]); o[1] = f2bf(v[1]);
                o[2] = f2bf(v[2]); o[3] = f2bf(v[3]);
                *(u16x4*)&As[row * LDA + lc] = o;
            }
            f32x4 wv = *(const f32x4*)(W + (size_t)(n0 + row) * K + k0 + lc);
            u16x4 ow;
            ow[0] = f2bf(wv[0]); ow[1] = f2bf(wv[1]);
            ow[2] = f2bf(wv[2]); ow[3] = f2bf(wv[3]);
            *(u16x4*)&Ws[row * LDA + lc] = ow;
        }
        __syncthreads();

        // ---- A fragments: lane<16 -> K 0..7 & 16..23, lane>=16 -> +8 ----
        const int kba = (lane < 16) ? 0 : 8;
        V16 afrag[2];
#pragma unroll
        for (int mt = 0; mt < 2; ++mt) {
            const int base = (wm + mt * 16 + (lane & 15)) * LDA + kba;
            afrag[mt].h[0] = *(const u16x8*)&As[base];
            afrag[mt].h[1] = *(const u16x8*)&As[base + 16];
        }
        // ---- B fragments: lane<16 -> K 0..7 & 16..23, lane>=16 -> +8 ----
        const int kbb = (lane < 16) ? 0 : 8;
        V16 bfrag[4];
#pragma unroll
        for (int nt = 0; nt < 4; ++nt) {
            const int base = (wn + nt * 16 + (lane & 15)) * LDA + kbb;
            bfrag[nt].h[0] = *(const u16x8*)&Ws[base];
            bfrag[nt].h[1] = *(const u16x8*)&Ws[base + 16];
        }
#pragma unroll
        for (int mt = 0; mt < 2; ++mt)
#pragma unroll
            for (int nt = 0; nt < 4; ++nt)
                acc[mt][nt] = wmma_bf16(afrag[mt].bf, bfrag[nt].bf, acc[mt][nt]);
    }

    // ---- epilogue: C layout -> vgpr r holds row (r + 8*(lane>=16)), col = lane&15
    const int mhalf = (lane >> 4) * 8;
    const int nlane = lane & 15;
#pragma unroll
    for (int mt = 0; mt < 2; ++mt) {
#pragma unroll
        for (int nt = 0; nt < 4; ++nt) {
            const int nn = n0 + wn + nt * 16 + nlane;
            const float bn = bias[nn];
#pragma unroll
            for (int r = 0; r < 8; ++r) {
                const int mm = m0 + wm + mt * 16 + mhalf + r;
                const float v = acc[mt][nt][r] + bn;
                if (MODE == 3) {
                    ofp[(size_t)mm * N + nn] = v;
                } else {
                    const int b = mm >> 11, s = mm & (SEQ - 1);
                    const int h = nn >> 6,  d = nn & (D_K - 1);
                    if (MODE == 0 || MODE == 1)
                        obf[((size_t)(b * NUM_HEADS + h) * SEQ + s) * D_K + d] = f2bf(v);
                    else // MODE 2: V transposed [b,h,d,s]
                        obf[((size_t)(b * NUM_HEADS + h) * D_K + d) * SEQ + s] = f2bf(v);
                }
            }
        }
    }
}

// ---------------------------------------------------------------------------
// Flash attention per (b,h): each wave owns 16 q rows, online softmax over
// 64-key blocks staged in LDS. scores = (Q K^T)/8 ; O = softmax(scores) V
// Q,K: [bh, S, 64] bf16.  Vt: [bh, 64, S] bf16.  O: [b*S, 1024] bf16.
// ---------------------------------------------------------------------------
__global__ __launch_bounds__(256) void attn_kernel(
    const unsigned short* __restrict__ Q, const unsigned short* __restrict__ Kh,
    const unsigned short* __restrict__ Vt, unsigned short* __restrict__ O)
{
    constexpr int LDK = 72;                        // padded LDS row stride
    __shared__ unsigned short Ks[64 * LDK];
    __shared__ unsigned short Vs[64 * LDK];
    __shared__ unsigned short Ps[8][16 * LDK];

    const int tid  = threadIdx.x;
    const int lane = tid & 31;
    const int wave = tid >> 5;
    const int bh = blockIdx.y;
    const int q0 = blockIdx.x * 128;
    const unsigned short* Qb = Q  + (size_t)bh * SEQ * D_K;
    const unsigned short* Kb = Kh + (size_t)bh * SEQ * D_K;
    const unsigned short* Vb = Vt + (size_t)bh * D_K * SEQ;

    const int qw0 = q0 + wave * 16;
    const int kba = (lane < 16) ? 0 : 8;           // A-fragment K base
    const int kbb = (lane < 16) ? 0 : 8;           // B-fragment K base

    // ---- Q fragments for this wave: 16 rows x 64 d, 2 K-steps of 32 ----
    V16 qf[2];
    {
        const int qrow = qw0 + (lane & 15);
#pragma unroll
        for (int ks2 = 0; ks2 < 2; ++ks2) {
            const int base = qrow * D_K + ks2 * 32 + kba;
            qf[ks2].h[0] = *(const u16x8*)(Qb + base);
            qf[ks2].h[1] = *(const u16x8*)(Qb + base + 16);
        }
    }

    float mrun[8], lrun[8];
    v8f oacc[4];
#pragma unroll
    for (int r = 0; r < 8; ++r) { mrun[r] = -INFINITY; lrun[r] = 0.0f; }
#pragma unroll
    for (int dt = 0; dt < 4; ++dt) oacc[dt] = (v8f)0.0f;

    const int lrow = tid >> 2;                     // 0..63 cooperative load row
    const int lcol = (tid & 3) * 16;               // 0,16,32,48
    const float LOG2E = 1.4426950408889634f;

    for (int k0 = 0; k0 < SEQ; k0 += 64) {
        __syncthreads();
        // ---- stage K block [64 keys x 64 d] and Vt block [64 d x 64 keys] ----
        {
            const u16x8* sk = (const u16x8*)(Kb + (size_t)(k0 + lrow) * D_K + lcol);
            *(u16x8*)&Ks[lrow * LDK + lcol]     = sk[0];
            *(u16x8*)&Ks[lrow * LDK + lcol + 8] = sk[1];
            const u16x8* sv = (const u16x8*)(Vb + (size_t)lrow * SEQ + k0 + lcol);
            *(u16x8*)&Vs[lrow * LDK + lcol]     = sv[0];
            *(u16x8*)&Vs[lrow * LDK + lcol + 8] = sv[1];
        }
        if (k0 + 64 < SEQ) {                       // global_prefetch_b8 next block
            __builtin_prefetch(Kb + (size_t)(k0 + 64 + lrow) * D_K + lcol, 0, 0);
            __builtin_prefetch(Vb + (size_t)lrow * SEQ + k0 + 64 + lcol, 0, 0);
        }
        __syncthreads();

        // ---- scores: 16 q x 64 keys = 4 tiles, K-dim = d (2 steps of 32) ----
        v8f sacc[4];
#pragma unroll
        for (int nt = 0; nt < 4; ++nt) sacc[nt] = (v8f)0.0f;
#pragma unroll
        for (int nt = 0; nt < 4; ++nt) {
            const int col = nt * 16 + (lane & 15);  // key index within block
#pragma unroll
            for (int ks2 = 0; ks2 < 2; ++ks2) {
                V16 bk;
                const int base = col * LDK + ks2 * 32 + kbb;
                bk.h[0] = *(const u16x8*)&Ks[base];
                bk.h[1] = *(const u16x8*)&Ks[base + 16];
                sacc[nt] = wmma_bf16(qf[ks2].bf, bk.bf, sacc[nt]);
            }
        }

        // ---- online softmax (rows split: lanes 0-15 = rows 0..7, 16-31 = 8..15)
        float bmax[8];
#pragma unroll
        for (int r = 0; r < 8; ++r) bmax[r] = -INFINITY;
#pragma unroll
        for (int nt = 0; nt < 4; ++nt)
#pragma unroll
            for (int r = 0; r < 8; ++r) {
                const float v = sacc[nt][r] * 0.125f;   // 1/sqrt(64)
                sacc[nt][r] = v;
                bmax[r] = fmaxf(bmax[r], v);
            }
#pragma unroll
        for (int r = 0; r < 8; ++r) {
            float x = bmax[r];
#pragma unroll
            for (int off = 1; off < 16; off <<= 1) x = fmaxf(x, __shfl_xor(x, off, 32));
            bmax[r] = x;
        }
        float fac[8];
#pragma unroll
        for (int r = 0; r < 8; ++r) {
            const float nm = fmaxf(mrun[r], bmax[r]);
            fac[r] = exp2f((mrun[r] - nm) * LOG2E);
            mrun[r] = nm;
        }
        float rsum[8];
#pragma unroll
        for (int r = 0; r < 8; ++r) rsum[r] = 0.0f;
#pragma unroll
        for (int nt = 0; nt < 4; ++nt)
#pragma unroll
            for (int r = 0; r < 8; ++r) {
                const float p = exp2f((sacc[nt][r] - mrun[r]) * LOG2E);
                sacc[nt][r] = p;
                rsum[r] += p;
            }
#pragma unroll
        for (int r = 0; r < 8; ++r) {
            float x = rsum[r];
#pragma unroll
            for (int off = 1; off < 16; off <<= 1) x += __shfl_xor(x, off, 32);
            lrun[r] = lrun[r] * fac[r] + x;
        }
#pragma unroll
        for (int dt = 0; dt < 4; ++dt)
#pragma unroll
            for (int r = 0; r < 8; ++r) oacc[dt][r] *= fac[r];

        // ---- re-layout P: C-format -> LDS row-major -> A-fragments ----
        unsigned short* Pw = &Ps[wave][0];
        const int mh = (lane >> 4) << 3;
#pragma unroll
        for (int nt = 0; nt < 4; ++nt)
#pragma unroll
            for (int r = 0; r < 8; ++r)
                Pw[(r + mh) * LDK + nt * 16 + (lane & 15)] = f2bf(sacc[nt][r]);
        asm volatile("s_wait_dscnt 0" ::: "memory");   // P visible before reload

        // ---- O += P(16x64) @ V(64x64): K-dim = key (2 steps), 4 d-tiles ----
#pragma unroll
        for (int dt = 0; dt < 4; ++dt) {
            const int colv = dt * 16 + (lane & 15);
#pragma unroll
            for (int ks2 = 0; ks2 < 2; ++ks2) {
                V16 ap, bv;
                const int basep = (lane & 15) * LDK + ks2 * 32 + kba;
                ap.h[0] = *(const u16x8*)&Pw[basep];
                ap.h[1] = *(const u16x8*)&Pw[basep + 16];
                const int basev = colv * LDK + ks2 * 32 + kbb;
                bv.h[0] = *(const u16x8*)&Vs[basev];
                bv.h[1] = *(const u16x8*)&Vs[basev + 16];
                oacc[dt] = wmma_bf16(ap.bf, bv.bf, oacc[dt]);
            }
        }
    }

    // ---- finalize: O[b, q, h*64+d] = oacc / l ----
    const int b = bh >> 4, h = bh & 15;
#pragma unroll
    for (int dt = 0; dt < 4; ++dt) {
        const int d = dt * 16 + (lane & 15);
#pragma unroll
        for (int r = 0; r < 8; ++r) {
            const int m = r + ((lane >> 4) << 3);
            const int qq = qw0 + m;
            const float v = oacc[dt][r] / lrun[r];
            O[((size_t)(b * SEQ + qq)) * D_MODEL + h * D_K + d] = f2bf(v);
        }
    }
}

// ---------------------------------------------------------------------------
extern "C" void kernel_launch(void* const* d_in, const int* in_sizes, int n_in,
                              void* d_out, int out_size, void* d_ws, size_t ws_size,
                              hipStream_t stream) {
    const float* q    = (const float*)d_in[0];
    const float* k    = (const float*)d_in[1];
    const float* v    = (const float*)d_in[2];
    const float* wq_w = (const float*)d_in[3];
    const float* wq_b = (const float*)d_in[4];
    const float* wk_w = (const float*)d_in[5];
    const float* wk_b = (const float*)d_in[6];
    const float* wv_w = (const float*)d_in[7];
    const float* wv_b = (const float*)d_in[8];
    const float* wo_w = (const float*)d_in[9];
    const float* wo_b = (const float*)d_in[10];

    const size_t ELTS = (size_t)MROWS * D_MODEL;       // 4M bf16 elements each
    unsigned short* qws = (unsigned short*)d_ws;
    unsigned short* kws = qws + ELTS;
    unsigned short* vws = kws + ELTS;
    unsigned short* ows = vws + ELTS;

    dim3 gridP(D_MODEL / 128, MROWS / 128);            // (8, 32)
    dim3 blk(256);
    proj_gemm<0><<<gridP, blk, 0, stream>>>(q, nullptr, wq_w, wq_b, qws, nullptr);
    proj_gemm<1><<<gridP, blk, 0, stream>>>(k, nullptr, wk_w, wk_b, kws, nullptr);
    proj_gemm<2><<<gridP, blk, 0, stream>>>(v, nullptr, wv_w, wv_b, vws, nullptr);

    dim3 gridA(SEQ / 128, BATCH * NUM_HEADS);          // (16, 32)
    attn_kernel<<<gridA, blk, 0, stream>>>(qws, kws, vws, ows);

    proj_gemm<3><<<gridP, blk, 0, stream>>>(nullptr, ows, wo_w, wo_b,
                                            nullptr, (float*)d_out);
}
